// MultiHeadSelfAttention2D_6579889898115
// MI455X (gfx1250) — matrix operand order repaired and checked by reading, not compiled, not run on hardware
//
#include <hip/hip_runtime.h>
#include <hip/hip_bf16.h>

typedef _Float16 h16;
typedef __attribute__((ext_vector_type(8)))  _Float16 v8h;
typedef __attribute__((ext_vector_type(16))) _Float16 v16h;
typedef __attribute__((ext_vector_type(8)))  float    v8f;

__device__ __forceinline__ v16h cat16(v8h lo, v8h hi) {
  v16h r;
#pragma unroll
  for (int t = 0; t < 8; ++t) { r[t] = lo[t]; r[t + 8] = hi[t]; }
  return r;
}

__device__ __forceinline__ v8f wmma16(v16h a, v16h b, v8f c) {
  return __builtin_amdgcn_wmma_f32_16x16x32_f16(false, a, false, b, (short)0, c,
                                                false, false);
}

// ---------------------------------------------------------------------------
// Prep: x [B,256,8,8] f32 -> Xh [B*64][256] f16  (row = b*64+s, col = c)
// LDS-tiled transpose.  The 64x64 source tile x[b][c0..c0+63][:] is one
// contiguous 16KB block -> stage it with CDNA5 async global->LDS B128 copies
// (ASYNCcnt), then emit converted, fully-coalesced 32B vector stores.
// ---------------------------------------------------------------------------
__global__ __launch_bounds__(256) void k_prep_x(const float* __restrict__ x,
                                                h16* __restrict__ Xh) {
  __shared__ __align__(16) float tile[64 * 64];  // [c][s], 16 KB
  const int t = threadIdx.x;
  const int b = blockIdx.x >> 2;
  const int c0 = (blockIdx.x & 3) * 64;
  const float* src = x + (size_t)b * 16384 + (size_t)c0 * 64;  // contiguous

  // async copy 16KB global -> LDS (16B per lane per issue, 4 issues)
  uint32_t lds0 = (uint32_t)(uintptr_t)(&tile[0]);  // low 32b = LDS offset
#pragma unroll
  for (int it = 0; it < 4; ++it) {
    int idx = it * 256 + t;                    // 16-byte chunk id
    uint32_t ldsa = lds0 + idx * 16;
    const float* g = src + idx * 4;
    asm volatile("global_load_async_to_lds_b128 %0, %1, off"
                 :: "v"(ldsa), "v"(g) : "memory");
  }
  asm volatile("s_wait_asynccnt 0" ::: "memory");
  __syncthreads();

  // transpose + convert: thread -> (s = t>>2, 16 consecutive c), 2x 16B stores
  int s = t >> 2;
  int cc0 = (t & 3) * 16;
  h16 outv[16];
#pragma unroll
  for (int u = 0; u < 16; ++u) outv[u] = (h16)tile[(cc0 + u) * 64 + s];
  h16* dst = Xh + ((size_t)b * 64 + s) * 256 + c0 + cc0;
  *(v8h*)(dst) = *(v8h*)&outv[0];
  *(v8h*)(dst + 8) = *(v8h*)&outv[8];
}

// ---------------------------------------------------------------------------
// Prep: transposed f16 weights + padded rel-pos embedding tables
//   WT [768][256]  rows 0-255 WqT, 256-511 WkT, 512-767 WvT  (WT[d][c]=W[c][d])
//   WoT[256][256]
//   Eh/Ew [16][16] f16, row 15 zero (vocab = 15)
// ---------------------------------------------------------------------------
__global__ __launch_bounds__(256) void k_prep_w(
    const float* __restrict__ Wq, const float* __restrict__ Wk,
    const float* __restrict__ Wv, const float* __restrict__ Wo,
    const float* __restrict__ eh, const float* __restrict__ ew,
    h16* __restrict__ WT, h16* __restrict__ WoT, h16* __restrict__ Eh,
    h16* __restrict__ Ew) {
  int idx = blockIdx.x * 256 + threadIdx.x;      // 262656 total
  if (idx < 196608) {
    int dp = idx >> 8, c = idx & 255;
    int sel = dp >> 8, d = dp & 255;
    const float* W = (sel == 0) ? Wq : (sel == 1) ? Wk : Wv;
    WT[idx] = (h16)W[c * 256 + d];
  } else if (idx < 262144) {
    int i2 = idx - 196608;
    int d = i2 >> 8, c = i2 & 255;
    WoT[i2] = (h16)Wo[c * 256 + d];
  } else if (idx < 262400) {
    int i3 = idx - 262144;
    int t = i3 >> 4, d = i3 & 15;
    Eh[i3] = (t < 15) ? (h16)eh[t * 16 + d] : (h16)0.f;
  } else if (idx < 262656) {
    int i4 = idx - 262400;
    int t = i4 >> 4, d = i4 & 15;
    Ew[i4] = (t < 15) ? (h16)ew[t * 16 + d] : (h16)0.f;
  }
}

// ---------------------------------------------------------------------------
// Fused QKV projection: [65536 x 256] x WT[768 x 256]^T, K=256.
// One wave = 16 rows x all 768 cols.  Outputs:
//   Qw[b][n][ij][32]  (scaled by 32^-0.5),  Kw same,  Vt[b][n][32][64]
// ---------------------------------------------------------------------------
__global__ __launch_bounds__(256) void k_proj(const h16* __restrict__ Xh,
                                              const h16* __restrict__ WT,
                                              h16* __restrict__ Qw,
                                              h16* __restrict__ Kw,
                                              h16* __restrict__ Vt) {
  const int lane = threadIdx.x & 31;
  const int l15 = lane & 15;
  const int hi16 = (lane >> 4) & 1;
  const int wid = blockIdx.x * 8 + (threadIdx.x >> 5);
  const int row0 = wid * 16;

  const h16* arow = Xh + (size_t)(row0 + l15) * 256;
  v8h alo[8], ahi[8];
#pragma unroll
  for (int kk = 0; kk < 8; ++kk) {
    int c1 = kk * 32 + hi16 * 8;
    alo[kk] = *(const v8h*)(arow + c1);
    ahi[kk] = *(const v8h*)(arow + c1 + 16);
  }

  for (int n0 = 0; n0 < 48; ++n0) {
    const h16* brow = WT + (size_t)(n0 * 16 + l15) * 256 + hi16 * 8;
    __builtin_prefetch(brow + 16 * 256, 0, 0);   // next n0's B row
    v8f acc = {};
#pragma unroll
    for (int kk = 0; kk < 8; ++kk) {
      v8h blo = *(const v8h*)(brow + kk * 32);
      v8h bhi = *(const v8h*)(brow + kk * 32 + 16);
      acc = wmma16(cat16(alo[kk], ahi[kk]), cat16(blo, bhi), acc);
    }
    int col = n0 * 16 + l15;                     // wave-uniform region
#pragma unroll
    for (int r = 0; r < 8; ++r) {
      int row = row0 + r + hi16 * 8;
      int b = row >> 6, ij = row & 63;
      float v = acc[r];
      if (col < 256) {
        int nh = col >> 5, d = col & 31;
        Qw[((size_t)(b * 8 + nh) * 64 + ij) * 32 + d] =
            (h16)(v * 0.17677669529663687f);
      } else if (col < 512) {
        int c2 = col - 256;
        int nh = c2 >> 5, d = c2 & 31;
        Kw[((size_t)(b * 8 + nh) * 64 + ij) * 32 + d] = (h16)v;
      } else {
        int c2 = col - 512;
        int nh = c2 >> 5, d = c2 & 31;
        Vt[((size_t)(b * 8 + nh) * 32 + d) * 64 + ij] = (h16)v;
      }
    }
  }
}

// ---------------------------------------------------------------------------
// Attention: one wave per (b, head).
//   logits = Q K^T (16 WMMA) + rel-pos (8 WMMA via LDS gather), f32 softmax,
//   attn = W x V (16 WMMA, W re-fragmented through LDS).
// ---------------------------------------------------------------------------
__global__ __launch_bounds__(32) void k_attn(const h16* __restrict__ Qw,
                                             const h16* __restrict__ Kw,
                                             const h16* __restrict__ Vt,
                                             const h16* __restrict__ Eh,
                                             const h16* __restrict__ Ew,
                                             h16* __restrict__ Cmb) {
  __shared__ float Ghl[64 * 16];
  __shared__ float Gwl[64 * 16];
  __shared__ __align__(16) h16 Wl[64 * 64];

  const int lane = threadIdx.x & 31;
  const int l15 = lane & 15;
  const int hi16 = (lane >> 4) & 1;
  const int bid = blockIdx.x;
  const int b = bid >> 3, nh = bid & 7;

  const h16* qb = Qw + (size_t)(b * 8 + nh) * 64 * 32;
  const h16* kb = Kw + (size_t)(b * 8 + nh) * 64 * 32;
  const h16* vb = Vt + (size_t)(b * 8 + nh) * 32 * 64;

  // Q as A-fragments (two 8-half chunks per lane)
  v8h qlo[4], qhi[4];
#pragma unroll
  for (int mi = 0; mi < 4; ++mi) {
    const h16* p = qb + (size_t)(mi * 16 + l15) * 32 + hi16 * 8;
    qlo[mi] = *(const v8h*)p;
    qhi[mi] = *(const v8h*)(p + 16);
  }
  // K as B-fragments (two 8-half chunks per lane, same k order as Q)
  v16h kf[4];
#pragma unroll
  for (int pi = 0; pi < 4; ++pi) {
    const h16* p = kb + (size_t)(pi * 16 + l15) * 32 + hi16 * 8;
    kf[pi] = cat16(*(const v8h*)p, *(const v8h*)(p + 16));
  }

  v8f acc[4][4];
#pragma unroll
  for (int mi = 0; mi < 4; ++mi)
#pragma unroll
    for (int pi = 0; pi < 4; ++pi) {
      v8f z = {};
      acc[mi][pi] = wmma16(cat16(qlo[mi], qhi[mi]), kf[pi], z);
    }

  // Rel-pos GEMMs: Gh = Q[:, :16] x Eh^T, Gw = Q[:, 16:] x Ew^T (K padded to 32)
  v16h ebh, ebw;
#pragma unroll
  for (int t = 0; t < 16; ++t) { ebh[t] = (h16)0.f; ebw[t] = (h16)0.f; }
  {
    v8h e0 = *(const v8h*)(Eh + l15 * 16 + hi16 * 8);
    v8h e1 = *(const v8h*)(Ew + l15 * 16 + hi16 * 8);
    for (int t = 0; t < 8; ++t) { ebh[t] = e0[t]; ebw[t] = e1[t]; }
  }
  v8h z8;
#pragma unroll
  for (int t = 0; t < 8; ++t) z8[t] = (h16)0.f;

#pragma unroll
  for (int mi = 0; mi < 4; ++mi) {
    v8f z = {};
    v8f gh = wmma16(cat16(qlo[mi], z8), ebh, z);
    v8f gw = wmma16(cat16(qhi[mi], z8), ebw, z);
#pragma unroll
    for (int r = 0; r < 8; ++r) {
      int row = mi * 16 + r + hi16 * 8;
      Ghl[row * 16 + l15] = gh[r];
      Gwl[row * 16 + l15] = gw[r];
    }
  }
  __syncthreads();

  // logits += Gh[ij, p-i+7] + Gw[ij, q-j+7]
#pragma unroll
  for (int mi = 0; mi < 4; ++mi)
#pragma unroll
    for (int pi = 0; pi < 4; ++pi) {
      int col = pi * 16 + l15;
      int p = col >> 3, q = col & 7;
#pragma unroll
      for (int r = 0; r < 8; ++r) {
        int row = mi * 16 + r + hi16 * 8;
        int i = row >> 3, j = row & 7;
        acc[mi][pi][r] += Ghl[row * 16 + (p - i + 7)] + Gwl[row * 16 + (q - j + 7)];
      }
    }

  // f32 softmax over the 64 keys (row spread across a 16-lane half x 4 frags)
#pragma unroll
  for (int mi = 0; mi < 4; ++mi) {
#pragma unroll
    for (int r = 0; r < 8; ++r) {
      float mx = fmaxf(fmaxf(acc[mi][0][r], acc[mi][1][r]),
                       fmaxf(acc[mi][2][r], acc[mi][3][r]));
      mx = fmaxf(mx, __shfl_xor(mx, 1));
      mx = fmaxf(mx, __shfl_xor(mx, 2));
      mx = fmaxf(mx, __shfl_xor(mx, 4));
      mx = fmaxf(mx, __shfl_xor(mx, 8));
      float s = 0.f;
#pragma unroll
      for (int pi = 0; pi < 4; ++pi) {
        float e = __expf(acc[mi][pi][r] - mx);
        acc[mi][pi][r] = e;
        s += e;
      }
      s += __shfl_xor(s, 1);
      s += __shfl_xor(s, 2);
      s += __shfl_xor(s, 4);
      s += __shfl_xor(s, 8);
      float inv = 1.0f / s;
#pragma unroll
      for (int pi = 0; pi < 4; ++pi) acc[mi][pi][r] *= inv;
    }
  }

  // Re-fragment weights: C-layout -> LDS -> A-layout
#pragma unroll
  for (int mi = 0; mi < 4; ++mi)
#pragma unroll
    for (int pi = 0; pi < 4; ++pi)
#pragma unroll
      for (int r = 0; r < 8; ++r) {
        int row = mi * 16 + r + hi16 * 8;
        Wl[row * 64 + pi * 16 + l15] = (h16)acc[mi][pi][r];
      }
  __syncthreads();

  // attn = W(64x64) x V(64x32)
#pragma unroll
  for (int mi = 0; mi < 4; ++mi) {
#pragma unroll
    for (int ni = 0; ni < 2; ++ni) {
      v8f o = {};
#pragma unroll
      for (int kh = 0; kh < 2; ++kh) {
        const h16* wp = &Wl[(mi * 16 + l15) * 64 + kh * 32 + hi16 * 8];
        v16h af = cat16(*(const v8h*)wp, *(const v8h*)(wp + 16));
        const h16* vp = vb + (size_t)(ni * 16 + l15) * 64 + kh * 32 + hi16 * 8;
        v16h bf = cat16(*(const v8h*)vp, *(const v8h*)(vp + 16));
        o = wmma16(af, bf, o);
      }
      int c = nh * 32 + ni * 16 + l15;
#pragma unroll
      for (int r = 0; r < 8; ++r) {
        int row = b * 64 + mi * 16 + r + hi16 * 8;
        Cmb[(size_t)row * 256 + c] = (h16)o[r];
      }
    }
  }
}

// ---------------------------------------------------------------------------
// Output projection: Cmb[65536 x 256] x WoT -> out f32 [B,256,8,8] (bdhw)
// ---------------------------------------------------------------------------
__global__ __launch_bounds__(256) void k_out(const h16* __restrict__ Cmb,
                                             const h16* __restrict__ WoT,
                                             float* __restrict__ Out) {
  const int lane = threadIdx.x & 31;
  const int l15 = lane & 15;
  const int hi16 = (lane >> 4) & 1;
  const int wid = blockIdx.x * 8 + (threadIdx.x >> 5);
  const int row0 = wid * 16;

  const h16* arow = Cmb + (size_t)(row0 + l15) * 256;
  v8h alo[8], ahi[8];
#pragma unroll
  for (int kk = 0; kk < 8; ++kk) {
    int c1 = kk * 32 + hi16 * 8;
    alo[kk] = *(const v8h*)(arow + c1);
    ahi[kk] = *(const v8h*)(arow + c1 + 16);
  }

  for (int n0 = 0; n0 < 16; ++n0) {
    const h16* brow = WoT + (size_t)(n0 * 16 + l15) * 256 + hi16 * 8;
    __builtin_prefetch(brow + 16 * 256, 0, 0);   // next n0's B row
    v8f acc = {};
#pragma unroll
    for (int kk = 0; kk < 8; ++kk) {
      v8h blo = *(const v8h*)(brow + kk * 32);
      v8h bhi = *(const v8h*)(brow + kk * 32 + 16);
      acc = wmma16(cat16(alo[kk], ahi[kk]), cat16(blo, bhi), acc);
    }
    int d = n0 * 16 + l15;
#pragma unroll
    for (int r = 0; r < 8; ++r) {
      int row = row0 + r + hi16 * 8;
      int b = row >> 6, s = row & 63;
      Out[(size_t)b * 16384 + d * 64 + s] = acc[r];
    }
  }
}

// ---------------------------------------------------------------------------
extern "C" void kernel_launch(void* const* d_in, const int* in_sizes, int n_in,
                              void* d_out, int out_size, void* d_ws,
                              size_t ws_size, hipStream_t stream) {
  (void)in_sizes; (void)n_in; (void)out_size; (void)ws_size;
  const float* x    = (const float*)d_in[0];
  const float* Wq   = (const float*)d_in[1];
  const float* Wk   = (const float*)d_in[2];
  const float* Wv   = (const float*)d_in[3];
  const float* Wo   = (const float*)d_in[4];
  const float* embh = (const float*)d_in[5];
  const float* embw = (const float*)d_in[6];

  const size_t SEG = 33554432;  // 65536*256 f16 bytes
  char* w = (char*)d_ws;
  h16* Xh  = (h16*)(w);
  h16* Qw  = (h16*)(w + SEG);
  h16* Kw  = (h16*)(w + 2 * SEG);
  h16* Vt  = (h16*)(w + 3 * SEG);
  h16* WT  = (h16*)(w + 4 * SEG);
  h16* WoT = WT + 768 * 256;
  h16* Eh  = WoT + 256 * 256;
  h16* Ew  = Eh + 256;
  h16* Cmb = Xh;  // Xh is dead after k_proj -> reuse for combined heads

  k_prep_x<<<4096, 256, 0, stream>>>(x, Xh);
  k_prep_w<<<1026, 256, 0, stream>>>(Wq, Wk, Wv, Wo, embh, embw, WT, WoT, Eh, Ew);
  k_proj<<<512, 256, 0, stream>>>(Xh, WT, Qw, Kw, Vt);
  k_attn<<<8192, 32, 0, stream>>>(Qw, Kw, Vt, Eh, Ew, Cmb);
  k_out<<<512, 256, 0, stream>>>(Cmb, WoT, (float*)d_out);
}
